// RecurrentTGCN_40037685133528
// MI455X (gfx1250) — matrix operand order repaired and checked by reading, not compiled, not run on hardware
//
#include <hip/hip_runtime.h>
#include <hip/hip_fp16.h>

typedef __attribute__((ext_vector_type(16))) _Float16 v16h;
typedef __attribute__((ext_vector_type(8)))  float    v8f;

#define NN   50000
#define EE   800000
#define HIDD 64
#define OUTD 16

// ---------------- graph normalization + aggregation ----------------

__global__ __launch_bounds__(256) void k_deg_init(float* __restrict__ deg) {
    int i = blockIdx.x * 256 + threadIdx.x;
    if (i < NN) deg[i] = 1.0f;                       // self-loop weight 1
}

__global__ __launch_bounds__(256) void k_deg_acc(float* __restrict__ deg,
                                                 const int* __restrict__ dst,
                                                 const float* __restrict__ ew) {
    int e = blockIdx.x * 256 + threadIdx.x;
    if (e < EE) atomicAdd(&deg[dst[e]], ew[e]);
}

__global__ __launch_bounds__(256) void k_dis(const float* __restrict__ deg,
                                             float* __restrict__ dis) {
    int i = blockIdx.x * 256 + threadIdx.x;
    if (i < NN) {
        float d = deg[i];
        dis[i] = (d > 0.0f) ? rsqrtf(fmaxf(d, 1e-12f)) : 0.0f;
    }
}

// aggx[i] = x[i] * dis[i]^2  (self-loop term), full init of aggx
__global__ __launch_bounds__(256) void k_self(const float* __restrict__ x,
                                              const float* __restrict__ dis,
                                              float* __restrict__ aggx) {
    int t = blockIdx.x * 256 + threadIdx.x;
    if (t < NN * HIDD) {
        int node = t >> 6;
        float s = dis[node];
        aggx[t] = x[t] * s * s;
    }
}

// one wave (32 lanes) per edge, 2 features per lane
__global__ __launch_bounds__(256) void k_edge(const float* __restrict__ x,
                                              const int* __restrict__ src,
                                              const int* __restrict__ dst,
                                              const float* __restrict__ ew,
                                              const float* __restrict__ dis,
                                              float* __restrict__ aggx) {
    int wave = threadIdx.x >> 5;
    int lane = threadIdx.x & 31;
    int e = blockIdx.x * 8 + wave;
    if (e >= EE) return;
    int s = src[e], d = dst[e];
    float coef = dis[s] * ew[e] * dis[d];
    const float2 v = ((const float2*)(x + (size_t)s * HIDD))[lane];
    float* ap = aggx + (size_t)d * HIDD + 2 * lane;
    atomicAdd(ap + 0, v.x * coef);
    atomicAdd(ap + 1, v.y * coef);
}

// ---------------- weight prep: fusion + WMMA-layout packing ----------------

// A = W @ Ltop   (64x64 = 64x64 @ 64x64; Ltop is first 64 rows of a 128x64)
__global__ __launch_bounds__(256) void k_mm64(const float* __restrict__ W,
                                              const float* __restrict__ Ltop,
                                              float* __restrict__ A) {
    int t = blockIdx.x * 256 + threadIdx.x;
    if (t < 64 * 64) {
        int i = t >> 6, j = t & 63;
        float s = 0.0f;
        for (int k = 0; k < 64; ++k) s += W[i * 64 + k] * Ltop[k * 64 + j];
        A[t] = s;
    }
}

// c[j] = b @ Ltop + Lb
__global__ __launch_bounds__(64) void k_biasfuse(const float* __restrict__ b,
                                                 const float* __restrict__ Ltop,
                                                 const float* __restrict__ Lb,
                                                 float* __restrict__ c) {
    int j = threadIdx.x;
    float s = Lb[j];
    for (int k = 0; k < 64; ++k) s += b[k] * Ltop[k * 64 + j];
    c[j] = s;
}

// Pack row-major KxNcol f32 into WMMA B-operand (32x16 f16) lane layout:
// dst[((t*ksteps+ks)*32 + lane)*16 + j] = src[(ks*32 + (lane>>4)*8 + (j&7) + (j>>3)*16)*Ncol + t*16 + (lane&15)]
__global__ __launch_bounds__(256) void k_pack(const float* __restrict__ src,
                                              _Float16* __restrict__ dst,
                                              int K, int Ncol) {
    int tid = blockIdx.x * 256 + threadIdx.x;
    int ksteps = K / 32;
    int total = (Ncol / 16) * ksteps * 512;
    if (tid < total) {
        int j    = tid & 15;
        int lane = (tid >> 4) & 31;
        int blk  = tid >> 9;
        int t    = blk / ksteps;
        int ks   = blk % ksteps;
        int hi   = lane >> 4, c = lane & 15;
        dst[tid] = (_Float16)src[(ks * 32 + hi * 8 + (j & 7) + (j >> 3) * 16) * Ncol + t * 16 + c];
    }
}

// ---------------- dense TGCN cell with WMMA ----------------

// Build 16-bit A-operand (16x32 MxK) for K-step k0 from a row-major f32 row:
// lanes 0-15 hold K = k0+[0..7] and k0+16+[0..7]; lanes 16-31 shifted by 8.
__device__ __forceinline__ v16h load_a_f32(const float* rowp, int k0, int hi) {
    const float* p = rowp + k0 + hi * 8;
    v16h a;
#pragma unroll
    for (int j = 0; j < 8; ++j) {
        a[j]     = (_Float16)p[j];
        a[j + 8] = (_Float16)p[16 + j];
    }
    return a;
}

__device__ __forceinline__ v16h load_a_lds(const _Float16* rowp, int k0, int hi) {
    const _Float16* p = rowp + k0 + hi * 8;
    v16h a;
#pragma unroll
    for (int j = 0; j < 8; ++j) {
        a[j]     = p[j];
        a[j + 8] = p[16 + j];
    }
    return a;
}

__device__ __forceinline__ v8f wmma_f16(v16h a, const _Float16* bp, v8f c) {
    v16h b = *(const v16h*)bp;
    return __builtin_amdgcn_wmma_f32_16x16x32_f16(false, a, false, b,
                                                  (short)0, c, false, false);
}

__global__ __launch_bounds__(256) void k_dense(
    const float* __restrict__ aggx, const float* __restrict__ Hprev,
    const _Float16* __restrict__ Pz, const _Float16* __restrict__ Qz, const float* __restrict__ cz,
    const _Float16* __restrict__ Pr, const _Float16* __restrict__ Qr, const float* __restrict__ cr,
    const _Float16* __restrict__ Ph, const _Float16* __restrict__ Qh, const float* __restrict__ ch,
    const _Float16* __restrict__ Plin, const float* __restrict__ yb,
    float* __restrict__ y, float* __restrict__ hout)
{
    __shared__ _Float16 lds_hr[8][16 * HIDD];   // H * R, f16, row-major per wave
    __shared__ _Float16 lds_eh[8][16 * HIDD];   // elu(h), f16, row-major per wave

    const int lane = threadIdx.x & 31;
    const int wave = threadIdx.x >> 5;
    const int hi   = lane >> 4;
    const int r    = lane & 15;        // A-row within tile == C-column within tile
    const int nb   = (blockIdx.x * 8 + wave) * 16;

    int arow = nb + r; if (arow > NN - 1) arow = NN - 1;
    const float* gx = aggx  + (size_t)arow * HIDD;
    const float* hx = Hprev + (size_t)arow * HIDD;

    v16h a_g0 = load_a_f32(gx, 0,  hi);
    v16h a_g1 = load_a_f32(gx, 32, hi);
    v16h a_h0 = load_a_f32(hx, 0,  hi);
    v16h a_h1 = load_a_f32(hx, 32, hi);

    float Hc[4][8];   // H in C-layout, kept for the final combine

    // ---- reset gate R, stage H*R in A-friendly row-major f16 ----
#pragma unroll
    for (int t = 0; t < 4; ++t) {
        v8f acc = {};
        acc = wmma_f16(a_g0, Pr + (t * 2 + 0) * 512 + lane * 16, acc);
        acc = wmma_f16(a_g1, Pr + (t * 2 + 1) * 512 + lane * 16, acc);
        acc = wmma_f16(a_h0, Qr + (t * 2 + 0) * 512 + lane * 16, acc);
        acc = wmma_f16(a_h1, Qr + (t * 2 + 1) * 512 + lane * 16, acc);
        float bias = cr[t * 16 + r];
#pragma unroll
        for (int g = 0; g < 8; ++g) {
            int row  = g + hi * 8;
            int node = nb + row; int cn = node > NN - 1 ? NN - 1 : node;
            float Hv = Hprev[(size_t)cn * HIDD + t * 16 + r];
            Hc[t][g] = Hv;
            float Rv = 1.0f / (1.0f + expf(-(acc[g] + bias)));
            lds_hr[wave][row * HIDD + t * 16 + r] = (_Float16)(Hv * Rv);
        }
    }
    __syncthreads();

    v16h a_q0 = load_a_lds(&lds_hr[wave][r * HIDD], 0,  hi);
    v16h a_q1 = load_a_lds(&lds_hr[wave][r * HIDD], 32, hi);

    // ---- update gate Z, candidate Ht, combine, stage elu(h) ----
#pragma unroll
    for (int t = 0; t < 4; ++t) {
        v8f accZ = {};
        accZ = wmma_f16(a_g0, Pz + (t * 2 + 0) * 512 + lane * 16, accZ);
        accZ = wmma_f16(a_g1, Pz + (t * 2 + 1) * 512 + lane * 16, accZ);
        accZ = wmma_f16(a_h0, Qz + (t * 2 + 0) * 512 + lane * 16, accZ);
        accZ = wmma_f16(a_h1, Qz + (t * 2 + 1) * 512 + lane * 16, accZ);
        v8f accH = {};
        accH = wmma_f16(a_g0, Ph + (t * 2 + 0) * 512 + lane * 16, accH);
        accH = wmma_f16(a_g1, Ph + (t * 2 + 1) * 512 + lane * 16, accH);
        accH = wmma_f16(a_q0, Qh + (t * 2 + 0) * 512 + lane * 16, accH);
        accH = wmma_f16(a_q1, Qh + (t * 2 + 1) * 512 + lane * 16, accH);
        float bz_ = cz[t * 16 + r];
        float bh_ = ch[t * 16 + r];
#pragma unroll
        for (int g = 0; g < 8; ++g) {
            int row  = g + hi * 8;
            int node = nb + row;
            float Zv  = 1.0f / (1.0f + expf(-(accZ[g] + bz_)));
            float Htv = tanhf(accH[g] + bh_);
            float hv  = Zv * Hc[t][g] + (1.0f - Zv) * Htv;
            if (node < NN) hout[(size_t)node * HIDD + t * 16 + r] = hv;
            float e = hv > 0.0f ? hv : expm1f(hv);
            lds_eh[wave][row * HIDD + t * 16 + r] = (_Float16)e;
        }
    }
    __syncthreads();

    // ---- y = elu(h) @ Wlin + b ----
    v16h a_e0 = load_a_lds(&lds_eh[wave][r * HIDD], 0,  hi);
    v16h a_e1 = load_a_lds(&lds_eh[wave][r * HIDD], 32, hi);
    v8f accY = {};
    accY = wmma_f16(a_e0, Plin + 0 * 512 + lane * 16, accY);
    accY = wmma_f16(a_e1, Plin + 1 * 512 + lane * 16, accY);
    float by = yb[r];
#pragma unroll
    for (int g = 0; g < 8; ++g) {
        int row  = g + hi * 8;
        int node = nb + row;
        if (node < NN) y[(size_t)node * OUTD + r] = accY[g] + by;
    }
}

// ---------------- host-side launch ----------------

extern "C" void kernel_launch(void* const* d_in, const int* in_sizes, int n_in,
                              void* d_out, int out_size, void* d_ws, size_t ws_size,
                              hipStream_t stream) {
    const float* x     = (const float*)d_in[0];
    const int*   ei    = (const int*)  d_in[1];   // [2][E]
    const float* ew    = (const float*)d_in[2];
    const float* Hprev = (const float*)d_in[3];
    const float* Wz    = (const float*)d_in[4];
    const float* bz    = (const float*)d_in[5];
    const float* Wr    = (const float*)d_in[6];
    const float* br    = (const float*)d_in[7];
    const float* Wh    = (const float*)d_in[8];
    const float* bh    = (const float*)d_in[9];
    const float* LzW   = (const float*)d_in[10];  // 128x64
    const float* Lzb   = (const float*)d_in[11];
    const float* LrW   = (const float*)d_in[12];
    const float* Lrb   = (const float*)d_in[13];
    const float* LhW   = (const float*)d_in[14];
    const float* Lhb   = (const float*)d_in[15];
    const float* WlinW = (const float*)d_in[16];  // 64x16
    const float* Wlinb = (const float*)d_in[17];

    float* y    = (float*)d_out;               // N x 16
    float* hout = (float*)d_out + (size_t)NN * OUTD;

    char* ws = (char*)d_ws;
    float*    aggx = (float*)(ws);                              // N*64 f32
    float*    deg  = (float*)(ws + 12800000);                   // N f32
    float*    dis  = (float*)(ws + 13000000);                   // N f32
    float*    Afz  = (float*)(ws + 13200000);                   // 3 x 64x64 f32 fused
    float*    cvec = (float*)(ws + 13200000 + 3 * 16384);       // 3 x 64 f32 fused bias
    _Float16* pk   = (_Float16*)(ws + 13200000 + 3 * 16384 + 768); // packed f16 operands

    _Float16* Pz   = pk;            // 4096 halves each
    _Float16* Pr   = pk + 4096;
    _Float16* Ph   = pk + 8192;
    _Float16* Qz   = pk + 12288;
    _Float16* Qr   = pk + 16384;
    _Float16* Qh   = pk + 20480;
    _Float16* Plin = pk + 24576;    // 1024 halves

    const int* src = ei;
    const int* dst = ei + EE;

    // 1) normalization
    k_deg_init<<<(NN + 255) / 256, 256, 0, stream>>>(deg);
    k_deg_acc <<<(EE + 255) / 256, 256, 0, stream>>>(deg, dst, ew);
    k_dis     <<<(NN + 255) / 256, 256, 0, stream>>>(deg, dis);

    // 2) aggregate raw features once:  aggx = A_norm @ x
    k_self<<<(NN * HIDD + 255) / 256, 256, 0, stream>>>(x, dis, aggx);
    k_edge<<<(EE + 7) / 8, 256, 0, stream>>>(x, src, dst, ew, dis, aggx);

    // 3) fuse weights:  A* = W* @ L*_top ;  c* = b* @ L*_top + L*_b
    k_mm64<<<16, 256, 0, stream>>>(Wz, LzW, Afz);
    k_mm64<<<16, 256, 0, stream>>>(Wr, LrW, Afz + 4096);
    k_mm64<<<16, 256, 0, stream>>>(Wh, LhW, Afz + 8192);
    k_biasfuse<<<1, 64, 0, stream>>>(bz, LzW, Lzb, cvec);
    k_biasfuse<<<1, 64, 0, stream>>>(br, LrW, Lrb, cvec + 64);
    k_biasfuse<<<1, 64, 0, stream>>>(bh, LhW, Lhb, cvec + 128);

    // 4) pack B-operands into WMMA lane layout (f16)
    k_pack<<<16, 256, 0, stream>>>(Afz,            Pz,   64, 64);
    k_pack<<<16, 256, 0, stream>>>(Afz + 4096,     Pr,   64, 64);
    k_pack<<<16, 256, 0, stream>>>(Afz + 8192,     Ph,   64, 64);
    k_pack<<<16, 256, 0, stream>>>(LzW + 64 * 64,  Qz,   64, 64);  // bottom half
    k_pack<<<16, 256, 0, stream>>>(LrW + 64 * 64,  Qr,   64, 64);
    k_pack<<<16, 256, 0, stream>>>(LhW + 64 * 64,  Qh,   64, 64);
    k_pack<<< 4, 256, 0, stream>>>(WlinW,          Plin, 64, 16);

    // 5) fused dense TGCN cell (WMMA)
    k_dense<<<(NN + 127) / 128, 256, 0, stream>>>(
        aggx, Hprev,
        Pz, Qz, cvec,
        Pr, Qr, cvec + 64,
        Ph, Qh, cvec + 128,
        Plin, Wlinb,
        y, hout);

    (void)in_sizes; (void)n_in; (void)out_size; (void)ws_size;
}
